// TopKSelector_9680856285433
// MI455X (gfx1250) — matrix operand order repaired and checked by reading, not compiled, not run on hardware
//
#include <hip/hip_runtime.h>
#include <hip/hip_bf16.h>
#include <math.h>

#define BATCH 4
#define NCAND 16384
#define DIM   1024
#define HID   256
#define KSEL  1024
#define ROWS  (BATCH * NCAND)   // 65536 total rows

typedef __attribute__((ext_vector_type(16))) __bf16 v16bf;
typedef __attribute__((ext_vector_type(8)))  float  v8f;

// ---------------------------------------------------------------------------
// Kernel 0: transpose + convert W1 [DIM,HID] f32  ->  W1T [HID,DIM] bf16
// ---------------------------------------------------------------------------
__global__ void topk_prep_w1t(const float* __restrict__ W1, __bf16* __restrict__ w1t) {
    int idx = blockIdx.x * 256 + threadIdx.x;          // HID*DIM = 262144
    int c = idx >> 10;                                 // / DIM
    int k = idx & (DIM - 1);
    w1t[idx] = (__bf16)W1[k * HID + c];
}

// ---------------------------------------------------------------------------
// Kernel 1: fused GEMM scorer.  Block = 256 threads (8 waves), covers 128 rows.
//   wave = (row_group 0..1) x (col_group 0..3); each wave: 64 rows x 64 cols
//   via 4x4 WMMA tiles of v_wmma_f32_16x16x32_bf16.
//   Epilogue: s[row] = relu(h + b1) . W2 + b2  (lane butterfly + ds_add_f32)
// ---------------------------------------------------------------------------
__global__ void __launch_bounds__(256)
topk_score_gemm(const float* __restrict__ x, const __bf16* __restrict__ w1t,
                const float* __restrict__ b1, const float* __restrict__ W2,
                const float* __restrict__ b2, float* __restrict__ s_out) {
    __shared__ float s_lds[128];

    const int tid  = threadIdx.x;
    const int lane = tid & 31;
    const int wave = tid >> 5;
    const int row_group = wave >> 2;         // 0..1
    const int col_group = wave & 3;          // 0..3
    const int rbase = blockIdx.x * 128 + row_group * 64;
    const int cbase = col_group * 64;
    const int l15   = lane & 15;
    const int hi    = (lane >> 4);           // 0 for lanes 0-15, 1 for 16-31
    const int a_koff = hi * 16;              // A: lo half K0-15, hi half K16-31
    const int b_koff = hi * 16;              // B: lo half K0-15, hi half K16-31

    if (tid < 128) s_lds[tid] = b2[0];       // bias b2 seeds the accumulation

    v8f acc[4][4] = {};

    for (int kb = 0; kb < DIM; kb += 32) {
        // B fragments: col = cbase + c*16 + l15, K contiguous in W1T
        v16bf bfr[4];
#pragma unroll
        for (int c = 0; c < 4; ++c) {
            const __bf16* bp = w1t + (size_t)(cbase + c * 16 + l15) * DIM + kb + b_koff;
            bfr[c] = *(const v16bf*)bp;
        }
#pragma unroll
        for (int r = 0; r < 4; ++r) {
            // A fragment: row = rbase + r*16 + l15; convert f32 -> bf16
            const float* rp = x + (size_t)(rbase + r * 16 + l15) * DIM + kb + a_koff;
            float4 f0 = *(const float4*)(rp);
            float4 f1 = *(const float4*)(rp + 4);
            float4 f2 = *(const float4*)(rp + 8);
            float4 f3 = *(const float4*)(rp + 12);
            v16bf a;
            a[0]  = (__bf16)f0.x; a[1]  = (__bf16)f0.y; a[2]  = (__bf16)f0.z; a[3]  = (__bf16)f0.w;
            a[4]  = (__bf16)f1.x; a[5]  = (__bf16)f1.y; a[6]  = (__bf16)f1.z; a[7]  = (__bf16)f1.w;
            a[8]  = (__bf16)f2.x; a[9]  = (__bf16)f2.y; a[10] = (__bf16)f2.z; a[11] = (__bf16)f2.w;
            a[12] = (__bf16)f3.x; a[13] = (__bf16)f3.y; a[14] = (__bf16)f3.z; a[15] = (__bf16)f3.w;
#pragma unroll
            for (int c = 0; c < 4; ++c) {
                acc[r][c] = __builtin_amdgcn_wmma_f32_16x16x32_bf16(
                    false, a, false, bfr[c], (short)0, acc[r][c], false, false);
            }
        }
    }

    __syncthreads();   // s_lds init visible before ds_add_f32 below

    // epilogue: per row-tile, fold relu(h+b1)*W2 across this wave's 64 columns
#pragma unroll
    for (int r = 0; r < 4; ++r) {
        float part[8];
#pragma unroll
        for (int v = 0; v < 8; ++v) part[v] = 0.0f;
#pragma unroll
        for (int c = 0; c < 4; ++c) {
            const int col = cbase + c * 16 + l15;
            const float b1v = b1[col];
            const float w2v = W2[col];
#pragma unroll
            for (int v = 0; v < 8; ++v) {
                float h = acc[r][c][v] + b1v;
                h = fmaxf(h, 0.0f);
                part[v] += h * w2v;
            }
        }
        // butterfly within each 16-lane half (N is striped over lanes 0-15 / 16-31)
#pragma unroll
        for (int m = 1; m <= 8; m <<= 1) {
#pragma unroll
            for (int v = 0; v < 8; ++v) part[v] += __shfl_xor(part[v], m, 32);
        }
        // VGPR v: lanes 0-15 hold row r*16+v, lanes 16-31 hold row r*16+8+v
        if (lane == 0) {
#pragma unroll
            for (int v = 0; v < 8; ++v)
                atomicAdd(&s_lds[row_group * 64 + r * 16 + v], part[v]);
        } else if (lane == 16) {
#pragma unroll
            for (int v = 0; v < 8; ++v)
                atomicAdd(&s_lds[row_group * 64 + r * 16 + 8 + v], part[v]);
        }
    }
    __syncthreads();
    if (tid < 128) s_out[(size_t)blockIdx.x * 128 + tid] = s_lds[tid];
}

// ---------------------------------------------------------------------------
// Kernel 2: per-batch softmax stats (max, sum of exp) over N=16384
// ---------------------------------------------------------------------------
__global__ void __launch_bounds__(1024)
topk_softmax_stats(const float* __restrict__ s, float* __restrict__ smax,
                   float* __restrict__ ssum) {
    __shared__ float red[1024];
    const int b = blockIdx.x, tid = threadIdx.x;
    const float* sb = s + (size_t)b * NCAND;

    float m = -INFINITY;
    for (int i = tid; i < NCAND; i += 1024) m = fmaxf(m, sb[i]);
    red[tid] = m;
    __syncthreads();
    for (int off = 512; off > 0; off >>= 1) {
        if (tid < off) red[tid] = fmaxf(red[tid], red[tid + off]);
        __syncthreads();
    }
    m = red[0];
    __syncthreads();

    float sm = 0.0f;
    for (int i = tid; i < NCAND; i += 1024) sm += __expf(sb[i] - m);
    red[tid] = sm;
    __syncthreads();
    for (int off = 512; off > 0; off >>= 1) {
        if (tid < off) red[tid] += red[tid + off];
        __syncthreads();
    }
    if (tid == 0) { smax[b] = m; ssum[b] = red[0]; }
}

// ---------------------------------------------------------------------------
// Kernel 3: exact top-k by rank counting.  rank(i) = #{j : s_j > s_i or
// (s_j == s_i and j < i)} reproduces lax.top_k descending order + index
// tie-break; all ranks are unique so each output slot is written once.
// ---------------------------------------------------------------------------
__global__ void __launch_bounds__(256)
topk_rank_select(const float* __restrict__ s, const float* __restrict__ smax,
                 const float* __restrict__ ssum, float* __restrict__ scores,
                 int* __restrict__ indices) {
    __shared__ float tile[256];
    const int b   = blockIdx.x >> 6;            // NCAND/256 = 64 tiles per batch
    const int i   = (blockIdx.x & 63) * 256 + threadIdx.x;
    const float* sb = s + (size_t)b * NCAND;
    const float si  = sb[i];

    int rank = 0;
    for (int t = 0; t < NCAND; t += 256) {
        tile[threadIdx.x] = sb[t + threadIdx.x];
        __syncthreads();
#pragma unroll 8
        for (int j = 0; j < 256; ++j) {
            const float sj = tile[j];
            const int   jg = t + j;
            rank += (sj > si) || (sj == si && jg < i);
        }
        __syncthreads();
    }
    if (rank < KSEL) {
        scores[b * KSEL + rank]  = __expf(si - smax[b]) / ssum[b];
        indices[b * KSEL + rank] = i;
    }
}

// ---------------------------------------------------------------------------
// Kernel 4: gather selected rows: sel[b,r,:] = x[b, idx[b,r], :]  (float4)
// ---------------------------------------------------------------------------
__global__ void __launch_bounds__(256)
topk_gather(const float* __restrict__ x, const int* __restrict__ indices,
            float* __restrict__ sel) {
    const int b = blockIdx.x >> 10;             // / KSEL
    const int r = blockIdx.x & (KSEL - 1);
    const int id = indices[b * KSEL + r];
    const float4* src = (const float4*)(x + ((size_t)b * NCAND + id) * DIM);
    float4*       dst = (float4*)(sel + ((size_t)b * KSEL + r) * DIM);
    __builtin_prefetch(src + threadIdx.x, 0, 0);   // global_prefetch_b8
    dst[threadIdx.x] = src[threadIdx.x];           // 256 lanes x float4 = 1024 f32
}

// ---------------------------------------------------------------------------
extern "C" void kernel_launch(void* const* d_in, const int* in_sizes, int n_in,
                              void* d_out, int out_size, void* d_ws, size_t ws_size,
                              hipStream_t stream) {
    const float* x  = (const float*)d_in[0];
    const float* W1 = (const float*)d_in[1];
    const float* b1 = (const float*)d_in[2];
    const float* W2 = (const float*)d_in[3];
    const float* b2 = (const float*)d_in[4];
    // d_in[5] = k (constant 1024, baked in)

    // outputs: sel [B,K,DIM] f32, scores [B,K] f32, indices [B,K] i32 (bit-cast)
    float* sel    = (float*)d_out;
    float* scores = sel + (size_t)BATCH * KSEL * DIM;
    int*   idxout = (int*)(scores + BATCH * KSEL);

    // workspace layout
    char* ws = (char*)d_ws;
    __bf16* w1t  = (__bf16*)ws;                                  // 512 KB
    float*  s    = (float*)(ws + (size_t)HID * DIM * sizeof(__bf16));
    float*  smax = s + ROWS;                                     // 4 floats
    float*  ssum = smax + BATCH;                                 // 4 floats

    topk_prep_w1t<<<HID * DIM / 256, 256, 0, stream>>>(W1, w1t);
    topk_score_gemm<<<ROWS / 128, 256, 0, stream>>>(x, w1t, b1, W2, b2, s);
    topk_softmax_stats<<<BATCH, 1024, 0, stream>>>(s, smax, ssum);
    topk_rank_select<<<BATCH * (NCAND / 256), 256, 0, stream>>>(s, smax, ssum,
                                                                scores, idxout);
    topk_gather<<<BATCH * KSEL, 256, 0, stream>>>(x, idxout, sel);
}
